// GoldenMoEPHRouting_9981503995948
// MI455X (gfx1250) — matrix operand order repaired and checked by reading, not compiled, not run on hardware
//
#include <hip/hip_runtime.h>
#include <hip/hip_bf16.h>
#include <math.h>

// Problem sizes (fixed by the reference)
#define N_TOK 4096
#define D_DIM 2048
#define H_DIM 4096
#define O_DIM 2048
#define E_EXP 8
#define KTOP  5
#define AHID  32
#define PHD   16

typedef __attribute__((ext_vector_type(16))) __bf16 v16bf;
typedef __attribute__((ext_vector_type(8)))  __bf16 v8bf;
typedef __attribute__((ext_vector_type(8)))  float  v8f;
typedef unsigned short u16;
typedef int v4i __attribute__((vector_size(16)));
typedef __attribute__((address_space(1))) v4i* gv4i_p;   // global 16B vector ptr
typedef __attribute__((address_space(3))) v4i* lv4i_p;   // LDS 16B vector ptr

// GEMM tiling: 128x128 block tile, 4 waves (2x2), 64x64 wave tile, BK=32
#define BM 128
#define BN 128
#define BK 32
#define LDT 40             // padded LDS stride in elements (40*2 = 80B, mult of 16B)
#define TSTRIDE (BM * LDT) // one LDS tile buffer, in elements

#define HAS_ASYNC_LDS __has_builtin(__builtin_amdgcn_global_load_async_to_lds_b128)

// async 16B global->LDS copy (per lane), tracked by ASYNCcnt
__device__ __forceinline__ void async_cp16(const u16* g, __bf16* l) {
#if HAS_ASYNC_LDS
  __builtin_amdgcn_global_load_async_to_lds_b128((gv4i_p)g, (lv4i_p)l, 0, 0);
#else
  (void)g; (void)l;
#endif
}

__device__ __forceinline__ void wait_async_zero() {
#if __has_builtin(__builtin_amdgcn_s_wait_asynccnt)
  __builtin_amdgcn_s_wait_asynccnt(0);
#else
  asm volatile("s_wait_asynccnt 0x0" ::: "memory");
#endif
}

// ---------------------------------------------------------------------------
// Fragment loaders (CDNA5 ISA 7.12.2 16-bit layouts)
// ---------------------------------------------------------------------------
__device__ __forceinline__ v16bf load_a_frag(const __bf16* sh, int row0) {
  const int lane = threadIdx.x & 31;
  const int m  = lane & 15;
  const int kh = (lane >> 4) << 3;                 // 0 or 8
  const __bf16* p = sh + (row0 + m) * LDT + kh;
  v8bf lo = *(const v8bf*)(p);                     // K = kh .. kh+7
  v8bf hi = *(const v8bf*)(p + 16);                // K = kh+16 .. kh+23
  v16bf r;
#pragma unroll
  for (int i = 0; i < 8; ++i) { r[i] = lo[i]; r[i + 8] = hi[i]; }
  return r;
}

__device__ __forceinline__ v16bf load_b_frag(const __bf16* sh, int col0) {
  const int lane = threadIdx.x & 31;
  const int n  = lane & 15;
  const int ko = (lane >> 4) << 3;                 // 0 or 8
  const __bf16* p = sh + (col0 + n) * LDT + ko;
  v8bf lo = *(const v8bf*)(p);
  v8bf hi = *(const v8bf*)(p + 16);
  v16bf r;
#pragma unroll
  for (int i = 0; i < 8; ++i) { r[i] = lo[i]; r[i + 8] = hi[i]; }
  return r;
}

// ---------------------------------------------------------------------------
// Pack: fp32 -> bf16 elementwise (for x). 8 elements per thread.
// ---------------------------------------------------------------------------
__global__ __launch_bounds__(256)
void cvt_bf16_kernel(const float* __restrict__ in, u16* __restrict__ out, long n) {
  const long i = ((long)blockIdx.x * 256 + threadIdx.x) * 8;
  if (i + 8 > n) return;
  const float4 a = *(const float4*)(in + i);
  const float4 b = *(const float4*)(in + i + 4);
  __bf16 v[8];
  v[0] = (__bf16)a.x; v[1] = (__bf16)a.y; v[2] = (__bf16)a.z; v[3] = (__bf16)a.w;
  v[4] = (__bf16)b.x; v[5] = (__bf16)b.y; v[6] = (__bf16)b.z; v[7] = (__bf16)b.w;
  *(uint4*)(out + i) = *(const uint4*)v;
}

// ---------------------------------------------------------------------------
// Pack: transpose + convert weight matrix fp32 [K, Nd] -> bf16 [Nd, K]
// ---------------------------------------------------------------------------
__global__ __launch_bounds__(256)
void transpose_cvt_kernel(const float* __restrict__ in, u16* __restrict__ out,
                          int Kd, int Nd) {
  __shared__ __bf16 t[64][72];
  const int tid = threadIdx.x;
  const int n0 = blockIdx.x * 64;
  const int k0 = blockIdx.y * 64;
#pragma unroll
  for (int i = tid; i < 64 * 64; i += 256) {
    const int kk = i >> 6, nn = i & 63;
    t[kk][nn] = (__bf16)in[(size_t)(k0 + kk) * Nd + n0 + nn];
  }
  __syncthreads();
#pragma unroll
  for (int i = tid; i < 64 * 64; i += 256) {
    const int nn = i >> 6, kk = i & 63;
    __bf16 v = t[kk][nn];
    out[(size_t)(n0 + nn) * Kd + k0 + kk] = *(const u16*)&v;
  }
}

// ---------------------------------------------------------------------------
// Routing: probs * ph_match, top-5, renormalize  ->  wgt[N,E]
// ---------------------------------------------------------------------------
__global__ __launch_bounds__(256)
void routing_kernel(const float* __restrict__ x,  const float* __restrict__ Wg,
                    const float* __restrict__ bg, const float* __restrict__ Wa,
                    const float* __restrict__ ba, const float* __restrict__ Wb,
                    const float* __restrict__ bb, const float* __restrict__ sigs,
                    float* __restrict__ wgt) {
  const int n   = blockIdx.x;
  const int tid = threadIdx.x;
  __shared__ float red[E_EXP + AHID];
  if (tid < E_EXP + AHID) red[tid] = 0.f;
  __syncthreads();

  float ag[E_EXP] = {};
  float aa[AHID]  = {};
  const float* xr = x + (size_t)n * D_DIM;
  for (int d = tid; d < D_DIM; d += 256) {
    const float xv = xr[d];
#pragma unroll
    for (int e2 = 0; e2 < E_EXP; ++e2) ag[e2] += xv * Wg[d * E_EXP + e2];
#pragma unroll
    for (int j = 0; j < AHID; ++j)     aa[j]  += xv * Wa[d * AHID + j];
  }
#pragma unroll
  for (int off = 16; off > 0; off >>= 1) {
#pragma unroll
    for (int e2 = 0; e2 < E_EXP; ++e2) ag[e2] += __shfl_down(ag[e2], off, 32);
#pragma unroll
    for (int j = 0; j < AHID; ++j)     aa[j]  += __shfl_down(aa[j], off, 32);
  }
  if ((tid & 31) == 0) {
#pragma unroll
    for (int e2 = 0; e2 < E_EXP; ++e2) atomicAdd(&red[e2], ag[e2]);
#pragma unroll
    for (int j = 0; j < AHID; ++j)     atomicAdd(&red[E_EXP + j], aa[j]);
  }
  __syncthreads();

  if (tid == 0) {
    const float TEMP = 2.718281828459045f;
    float probs[E_EXP];
    float mx = -1e30f;
#pragma unroll
    for (int e2 = 0; e2 < E_EXP; ++e2) {
      probs[e2] = (red[e2] + bg[e2]) / TEMP;
      mx = fmaxf(mx, probs[e2]);
    }
    float s = 0.f;
#pragma unroll
    for (int e2 = 0; e2 < E_EXP; ++e2) { probs[e2] = __expf(probs[e2] - mx); s += probs[e2]; }
#pragma unroll
    for (int e2 = 0; e2 < E_EXP; ++e2) probs[e2] /= s;
    float a[AHID];
#pragma unroll
    for (int j = 0; j < AHID; ++j) { float v = red[E_EXP + j] + ba[j]; a[j] = v > 0.f ? v : 0.f; }
    float ph[PHD];
#pragma unroll
    for (int p = 0; p < PHD; ++p) {
      float v = bb[p];
#pragma unroll
      for (int j = 0; j < AHID; ++j) v += a[j] * Wb[j * PHD + p];
      ph[p] = v;
    }
    float nrm = 0.f;
#pragma unroll
    for (int p = 0; p < PHD; ++p) nrm += ph[p] * ph[p];
    nrm = fmaxf(sqrtf(nrm), 1e-12f);
    float eff[E_EXP];
#pragma unroll
    for (int e2 = 0; e2 < E_EXP; ++e2) {
      const float* sg = sigs + e2 * PHD;
      float sn = 0.f, dot = 0.f;
#pragma unroll
      for (int p = 0; p < PHD; ++p) sn += sg[p] * sg[p];
      sn = fmaxf(sqrtf(sn), 1e-12f);
#pragma unroll
      for (int p = 0; p < PHD; ++p) dot += (ph[p] / nrm) * (sg[p] / sn);
      eff[e2] = probs[e2] * (dot + 1.f) * 0.5f;
    }
    bool sel[E_EXP] = {};
    float wsum = 0.f;
    for (int t = 0; t < KTOP; ++t) {
      int bi = 0; float bv = -1e30f;
#pragma unroll
      for (int e2 = 0; e2 < E_EXP; ++e2)
        if (!sel[e2] && eff[e2] > bv) { bv = eff[e2]; bi = e2; }
      sel[bi] = true; wsum += eff[bi];
    }
    const float inv = 1.f / (wsum + 1e-8f);
#pragma unroll
    for (int e2 = 0; e2 < E_EXP; ++e2)
      wgt[(size_t)n * E_EXP + e2] = sel[e2] ? eff[e2] * inv : 0.f;
  }
}

// ---------------------------------------------------------------------------
// bf16 GEMM, all operands pre-packed bf16:
//   A [M,K] row-major (lda), Bt [N,K] row-major (ldb, i.e. B transposed)
// 128x128 block tile, 4 waves, 64x64 wave tile (16 WMMA / wave / K-step).
// Async global->LDS double buffering (ASYNCcnt) when available; otherwise
// register-staged prefetch.
// Epilogue: hout != null -> h = relu(acc + bias) as bf16
//           else          -> fout (+)= wgt[row,e] * (acc + bias)
// ---------------------------------------------------------------------------
__global__ __launch_bounds__(128, 1)
void gemm_bf16_kernel(const u16* __restrict__ Ag, int lda,
                      const u16* __restrict__ Bg, int ldb, int K,
                      const float* __restrict__ bias,
                      const float* __restrict__ wgt, int e, int first,
                      u16* __restrict__ hout, float* __restrict__ fout, int ldo) {
  __shared__ __bf16 ash[2 * TSTRIDE];
  __shared__ __bf16 bsh[2 * TSTRIDE];
  const int tid  = threadIdx.x;
  const int m0   = blockIdx.y * BM;
  const int n0   = blockIdx.x * BN;
  const int wave = tid >> 5;
  const int wm   = (wave & 1) * 64;
  const int wn   = (wave >> 1) * 64;

  v8f acc[4][4];
  const v8f vzero = {0.f, 0.f, 0.f, 0.f, 0.f, 0.f, 0.f, 0.f};
#pragma unroll
  for (int i = 0; i < 4; ++i)
#pragma unroll
    for (int j = 0; j < 4; ++j) acc[i][j] = vzero;

  // per-thread tile slots: 4x 16B for A and for B
  const int r_sl  = tid >> 2;          // row base 0..31
  const int kq_sl = (tid & 3) * 8;     // k offset 0,8,16,24
  const int nT = K / BK;

#if HAS_ASYNC_LDS
  // ------------------ async double-buffered pipeline --------------------
  // issue tile 0 into buffer 0
#pragma unroll
  for (int j = 0; j < 4; ++j) {
    const int r = r_sl + j * 32;
    async_cp16(Ag + (size_t)(m0 + r) * lda + kq_sl, ash + r * LDT + kq_sl);
    async_cp16(Bg + (size_t)(n0 + r) * ldb + kq_sl, bsh + r * LDT + kq_sl);
  }
  wait_async_zero();
  __syncthreads();

  for (int t = 0; t < nT; ++t) {
    const int buf = t & 1;
    if (t + 1 < nT) {
      const int k0 = (t + 1) * BK;
      const int nxt = (1 - buf) * TSTRIDE;
#pragma unroll
      for (int j = 0; j < 4; ++j) {
        const int r = r_sl + j * 32;
        async_cp16(Ag + (size_t)(m0 + r) * lda + k0 + kq_sl, ash + nxt + r * LDT + kq_sl);
        async_cp16(Bg + (size_t)(n0 + r) * ldb + k0 + kq_sl, bsh + nxt + r * LDT + kq_sl);
      }
    }
    const __bf16* as = ash + buf * TSTRIDE;
    const __bf16* bs = bsh + buf * TSTRIDE;
    v16bf bfrag[4];
#pragma unroll
    for (int fn = 0; fn < 4; ++fn) bfrag[fn] = load_b_frag(bs, wn + fn * 16);
#pragma unroll
    for (int fm = 0; fm < 4; ++fm) {
      v16bf afrag = load_a_frag(as, wm + fm * 16);
#pragma unroll
      for (int fn = 0; fn < 4; ++fn)
        acc[fm][fn] = __builtin_amdgcn_wmma_f32_16x16x32_bf16(
            false, afrag, false, bfrag[fn], (short)0, acc[fm][fn], false, false);
    }
    wait_async_zero();   // next tile's copies are complete for this wave
    __syncthreads();     // all waves done reading current buffer + waiting
  }
#else
  // ------------------ register-staged prefetch fallback -----------------
  uint4 pa[4], pb[4];
#pragma unroll
  for (int j = 0; j < 4; ++j) {
    const int r = r_sl + j * 32;
    pa[j] = *(const uint4*)(Ag + (size_t)(m0 + r) * lda + kq_sl);
    pb[j] = *(const uint4*)(Bg + (size_t)(n0 + r) * ldb + kq_sl);
  }
#pragma unroll
  for (int j = 0; j < 4; ++j) {
    const int r = r_sl + j * 32;
    *(uint4*)&ash[r * LDT + kq_sl] = pa[j];
    *(uint4*)&bsh[r * LDT + kq_sl] = pb[j];
  }
  __syncthreads();
  for (int t = 0; t < nT; ++t) {
    if (t + 1 < nT) {
      const int k0 = (t + 1) * BK;
#pragma unroll
      for (int j = 0; j < 4; ++j) {
        const int r = r_sl + j * 32;
        pa[j] = *(const uint4*)(Ag + (size_t)(m0 + r) * lda + k0 + kq_sl);
        pb[j] = *(const uint4*)(Bg + (size_t)(n0 + r) * ldb + k0 + kq_sl);
      }
    }
    v16bf bfrag[4];
#pragma unroll
    for (int fn = 0; fn < 4; ++fn) bfrag[fn] = load_b_frag(bsh, wn + fn * 16);
#pragma unroll
    for (int fm = 0; fm < 4; ++fm) {
      v16bf afrag = load_a_frag(ash, wm + fm * 16);
#pragma unroll
      for (int fn = 0; fn < 4; ++fn)
        acc[fm][fn] = __builtin_amdgcn_wmma_f32_16x16x32_bf16(
            false, afrag, false, bfrag[fn], (short)0, acc[fm][fn], false, false);
    }
    __syncthreads();
    if (t + 1 < nT) {
#pragma unroll
      for (int j = 0; j < 4; ++j) {
        const int r = r_sl + j * 32;
        *(uint4*)&ash[r * LDT + kq_sl] = pa[j];
        *(uint4*)&bsh[r * LDT + kq_sl] = pb[j];
      }
      __syncthreads();
    }
  }
#endif

  // epilogue (C frag: lanes 0-15 M=i, lanes 16-31 M=i+8; N = lane&15)
  const int lane = tid & 31;
  const int cn = lane & 15;
  const int cm = (lane >> 4) << 3;
  if (hout) {
#pragma unroll
    for (int fm = 0; fm < 4; ++fm)
#pragma unroll
      for (int fn = 0; fn < 4; ++fn)
#pragma unroll
        for (int i = 0; i < 8; ++i) {
          const int row = m0 + wm + fm * 16 + cm + i;
          const int col = n0 + wn + fn * 16 + cn;
          float v = acc[fm][fn][i] + bias[col];
          v = v > 0.f ? v : 0.f;
          __bf16 b = (__bf16)v;
          hout[(size_t)row * ldo + col] = *(const u16*)&b;
        }
  } else {
#pragma unroll
    for (int fm = 0; fm < 4; ++fm)
#pragma unroll
      for (int fn = 0; fn < 4; ++fn)
#pragma unroll
        for (int i = 0; i < 8; ++i) {
          const int row = m0 + wm + fm * 16 + cm + i;
          const int col = n0 + wn + fn * 16 + cn;
          const float w = wgt[(size_t)row * E_EXP + e];
          const float v = w * (acc[fm][fn][i] + bias[col]);
          const size_t idx = (size_t)row * ldo + col;
          fout[idx] = first ? v : (fout[idx] + v);
        }
  }
}

// ---------------------------------------------------------------------------
extern "C" void kernel_launch(void* const* d_in, const int* in_sizes, int n_in,
                              void* d_out, int out_size, void* d_ws, size_t ws_size,
                              hipStream_t stream) {
  (void)in_sizes; (void)n_in; (void)out_size; (void)ws_size;
  const float* x    = (const float*)d_in[0];
  const float* Wg   = (const float*)d_in[1];
  const float* bg   = (const float*)d_in[2];
  const float* Wa   = (const float*)d_in[3];
  const float* ba   = (const float*)d_in[4];
  const float* Wb   = (const float*)d_in[5];
  const float* bb   = (const float*)d_in[6];
  const float* sigs = (const float*)d_in[7];
  const float* W1   = (const float*)d_in[8];
  const float* b1   = (const float*)d_in[9];
  const float* W2   = (const float*)d_in[10];
  const float* b2   = (const float*)d_in[11];
  float* out = (float*)d_out;

  // workspace layout (bytes):
  //   [0,16M)    xbf  : x as bf16            N*D*2   = 16 MB
  //   [16M,48M)  h    : hidden bf16          N*H*2   = 32 MB
  //   [48M,64M)  Wt   : packed W^T bf16      max(H*D, O*H)*2 = 16 MB (reused)
  //   [64M,..)   wgt  : routing weights      N*E*4   = 128 KB
  char* ws = (char*)d_ws;
  u16*   xbf  = (u16*)ws;
  u16*   hbuf = (u16*)(ws + (size_t)N_TOK * D_DIM * 2);
  u16*   wt   = (u16*)(ws + (size_t)N_TOK * D_DIM * 2 + (size_t)N_TOK * H_DIM * 2);
  float* wgt  = (float*)(ws + (size_t)N_TOK * D_DIM * 2 + (size_t)N_TOK * H_DIM * 2
                            + (size_t)H_DIM * D_DIM * 2);

  routing_kernel<<<N_TOK, 256, 0, stream>>>(x, Wg, bg, Wa, ba, Wb, bb, sigs, wgt);
  {
    const long nx = (long)N_TOK * D_DIM;
    cvt_bf16_kernel<<<(unsigned)(nx / (256 * 8)), 256, 0, stream>>>(x, xbf, nx);
  }

  for (int e = 0; e < E_EXP; ++e) {
    // pack W1e^T: [D,H] fp32 -> [H,D] bf16
    transpose_cvt_kernel<<<dim3(H_DIM / 64, D_DIM / 64), 256, 0, stream>>>(
        W1 + (size_t)e * D_DIM * H_DIM, wt, D_DIM, H_DIM);
    // h = relu(xbf @ W1e + b1e)
    gemm_bf16_kernel<<<dim3(H_DIM / BN, N_TOK / BM), 128, 0, stream>>>(
        xbf, D_DIM, wt, D_DIM, D_DIM, b1 + (size_t)e * H_DIM,
        nullptr, e, 0, hbuf, nullptr, H_DIM);
    // pack W2e^T: [H,O] fp32 -> [O,H] bf16
    transpose_cvt_kernel<<<dim3(O_DIM / 64, H_DIM / 64), 256, 0, stream>>>(
        W2 + (size_t)e * H_DIM * O_DIM, wt, H_DIM, O_DIM);
    // out (+)= w[:,e] * (h @ W2e + b2e)
    gemm_bf16_kernel<<<dim3(O_DIM / BN, N_TOK / BM), 128, 0, stream>>>(
        hbuf, H_DIM, wt, H_DIM, H_DIM, b2 + (size_t)e * O_DIM,
        wgt, e, e == 0, nullptr, out, O_DIM);
  }
}
